// MiniGPT_37503654428854
// MI455X (gfx1250) — matrix operand order repaired and checked by reading, not compiled, not run on hardware
//
#include <hip/hip_runtime.h>

// ---------------------------------------------------------------------------
// MiniGPT forward for MI455X (gfx1250): bf16 WMMA GEMMs + flash attention,
// with async global->LDS tile movement (ASYNCcnt) and double-buffered GEMM.
// ---------------------------------------------------------------------------

typedef unsigned short u16;
typedef __attribute__((ext_vector_type(16))) __bf16 v16bf;
typedef __attribute__((ext_vector_type(8)))  float  v8f;
typedef __attribute__((ext_vector_type(4)))  unsigned int u32x4;

union Frag { v16bf v; u32x4 q[2]; };

#define DM    1024
#define FF    4096
#define SEQ   2048
#define NB    2
#define NH    16
#define DKC   64
#define NL    4
#define NV    50257
#define MROWS (NB * SEQ)      // 4096

__device__ __forceinline__ u16 f2bf(float f) {
  unsigned int u = __float_as_uint(f);
  u += 0x7FFFu + ((u >> 16) & 1u);    // round-to-nearest-even
  return (u16)(u >> 16);
}

// Async copy of 16 bytes per lane: global -> LDS, tracked by ASYNCcnt.
__device__ __forceinline__ void async_ld16(unsigned lds_off, const u16* g) {
  asm volatile("global_load_async_to_lds_b128 %0, %1, off"
               :: "v"(lds_off), "v"((unsigned long long)(size_t)g)
               : "memory");
}
__device__ __forceinline__ void wait_async0() {
  asm volatile("s_wait_asynccnt 0x0" ::: "memory");
}

// ---------------------------------------------------------------------------
// Elementwise fp32 -> bf16 conversion (weights & activations)
// ---------------------------------------------------------------------------
__global__ __launch_bounds__(256) void cvt_f32_bf16(
    const float* __restrict__ src, u16* __restrict__ dst, size_t n) {
  size_t stride = (size_t)gridDim.x * blockDim.x;
  for (size_t i = (size_t)blockIdx.x * blockDim.x + threadIdx.x; i < n; i += stride)
    dst[i] = f2bf(src[i]);
}

// ---------------------------------------------------------------------------
// Embedding lookup + sinusoidal positional encoding  -> h [B,S,D] fp32
// ---------------------------------------------------------------------------
__global__ __launch_bounds__(256) void embed_pe(
    const int* __restrict__ x, const float* __restrict__ emb, float* __restrict__ h) {
  const int total = MROWS * DM;
  for (int i = blockIdx.x * blockDim.x + threadIdx.x; i < total;
       i += gridDim.x * blockDim.x) {
    int row = i >> 10;             // b*S + s
    int d   = i & (DM - 1);
    int s   = row & (SEQ - 1);
    int tok = x[row];
    float divf = __expf((float)(d & ~1) * (-9.2103403719761836f / (float)DM));
    float ang  = (float)s * divf;
    float pe   = (d & 1) ? __cosf(ang) : __sinf(ang);
    h[i] = emb[(size_t)tok * DM + d] + pe;
  }
}

// ---------------------------------------------------------------------------
// LayerNorm over D=1024, fp32 in -> bf16 out. One 256-thread block per row.
// ---------------------------------------------------------------------------
__global__ __launch_bounds__(256) void ln_f32_bf16(
    const float* __restrict__ X, const float* __restrict__ g,
    const float* __restrict__ bb, u16* __restrict__ Y) {
  const int row = blockIdx.x, tid = threadIdx.x;
  const float* x = X + (size_t)row * DM;
  float s = 0.f, s2 = 0.f;
  for (int i = tid; i < DM; i += 256) { float v = x[i]; s += v; s2 += v * v; }
  __shared__ float rs[256], rs2[256];
  rs[tid] = s; rs2[tid] = s2;
  __syncthreads();
  for (int off = 128; off > 0; off >>= 1) {
    if (tid < off) { rs[tid] += rs[tid + off]; rs2[tid] += rs2[tid + off]; }
    __syncthreads();
  }
  float mu   = rs[0] * (1.0f / DM);
  float var  = rs2[0] * (1.0f / DM) - mu * mu;
  float rstd = rsqrtf(var + 1e-5f);
  u16* y = Y + (size_t)row * DM;
  for (int i = tid; i < DM; i += 256)
    y[i] = f2bf((x[i] - mu) * rstd * g[i] + bb[i]);
}

// ---------------------------------------------------------------------------
// Repack fp32 [B*S, D] -> bf16 [B, H, S, DK] for attention
// ---------------------------------------------------------------------------
__global__ __launch_bounds__(256) void repack_heads(
    const float* __restrict__ src, u16* __restrict__ dst) {
  const int total = MROWS * DM;
  for (int i = blockIdx.x * blockDim.x + threadIdx.x; i < total;
       i += gridDim.x * blockDim.x) {
    int d  = i & (DM - 1);
    int s  = (i >> 10) & (SEQ - 1);
    int b  = i >> 21;                       // / (D*S) = / 2^21
    int hh = d >> 6, dk = d & 63;
    dst[(((size_t)(b * NH + hh) * SEQ + s) << 6) + dk] = f2bf(src[i]);
  }
}

// ---------------------------------------------------------------------------
// Tiled bf16 GEMM:  C[M,N](fp32) = A[M,K](bf16,row-major) @ B[K,N](bf16,row-major)
//   + optional bias[N], + optional residual[M,N], + optional ReLU.
// 256 threads = 8 waves. Block tile 128x128, wave tile 32x64, K-step 32.
// Double-buffered LDS: A tile via global_load_async_to_lds_b128 (ASYNCcnt),
// B tile via software-pipelined VGPR transpose (needed for B-fragment layout).
// ---------------------------------------------------------------------------
__global__ __launch_bounds__(256) void gemm_bf16(
    const u16* __restrict__ A, const u16* __restrict__ Bm, float* __restrict__ C,
    const float* __restrict__ bias, const float* __restrict__ residual,
    int M, int N, int K, int relu) {
  __shared__ u16 As[2][128][40];   // [buf][m][k], padded
  __shared__ u16 Bs[2][128][40];   // [buf][n][k] (transposed), padded

  const int tid  = threadIdx.x;
  const int lane = tid & 31, wave = tid >> 5;
  const int wr = wave >> 1, wc = wave & 1;       // 4x2 wave grid
  const int hf = lane >> 4, ln15 = lane & 15;
  const int row0 = blockIdx.y * 128;
  const int col0 = blockIdx.x * 128;

  const int ra = tid >> 1, sa = (tid & 1) * 16;  // A-tile: 16 halfs / thread
  const int kb = tid >> 3, nb = (tid & 7) * 16;  // B-tile: 16 halfs / thread
  const bool fullN = (col0 + 128 <= N);

  u16 btmp[16];

  auto issueA = [&](int k0, int buf) {           // async global -> LDS
    const u16* src = A + (size_t)(row0 + ra) * K + k0 + sa;
    unsigned lds = (unsigned)(size_t)&As[buf][ra][sa];
    async_ld16(lds, src);
    async_ld16(lds + 16, src + 8);
  };
  auto loadB = [&](int k0) {                     // global -> VGPR
    const u16* src = Bm + (size_t)(k0 + kb) * N + col0 + nb;
    if (fullN) {
      u32x4 t0 = ((const u32x4*)src)[0];
      u32x4 t1 = ((const u32x4*)src)[1];
      const u16* p0 = (const u16*)&t0;
      const u16* p1 = (const u16*)&t1;
#pragma unroll
      for (int j = 0; j < 8; ++j) { btmp[j] = p0[j]; btmp[8 + j] = p1[j]; }
    } else {
#pragma unroll
      for (int j = 0; j < 16; ++j) {
        int c = col0 + nb + j;
        btmp[j] = (c < N) ? src[j] : (u16)0;
      }
    }
  };
  auto storeB = [&](int buf) {                   // VGPR -> LDS (transposed)
#pragma unroll
    for (int j = 0; j < 16; ++j) Bs[buf][nb + j][kb] = btmp[j];
  };

  v8f acc[2][4] = {};

  // prologue: stage tile 0
  issueA(0, 0);
  loadB(0);
  storeB(0);
  wait_async0();
  __syncthreads();

  const int iters = K >> 5;
  for (int it = 0; it < iters; ++it) {
    const int cur = it & 1;
    const bool more = (it + 1 < iters);
    if (more) {                      // prefetch next tile while computing
      issueA((it + 1) << 5, cur ^ 1);
      loadB((it + 1) << 5);
    }

    Frag aF[2], bF[4];
#pragma unroll
    for (int sub = 0; sub < 2; ++sub) {
      int r = wr * 32 + sub * 16 + ln15;               // A: lane holds row M
      aF[sub].q[0] = *(const u32x4*)&As[cur][r][8 * hf];      // K = 8*hf+0..7
      aF[sub].q[1] = *(const u32x4*)&As[cur][r][16 + 8 * hf]; // K = 16+8*hf+0..7
    }
#pragma unroll
    for (int t = 0; t < 4; ++t) {
      int n = wc * 64 + t * 16 + ln15;                 // B: lane holds column N
      bF[t].q[0] = *(const u32x4*)&Bs[cur][n][8 * hf];        // K = 8*hf+0..7
      bF[t].q[1] = *(const u32x4*)&Bs[cur][n][16 + 8 * hf];   // K = 16+8*hf+0..7
    }
#pragma unroll
    for (int sub = 0; sub < 2; ++sub)
#pragma unroll
      for (int t = 0; t < 4; ++t)
        acc[sub][t] = __builtin_amdgcn_wmma_f32_16x16x32_bf16(
            false, aF[sub].v, false, bF[t].v, (short)0, acc[sub][t], false, false);

    if (more) {
      storeB(cur ^ 1);    // LDS scatter lands after the WMMAs issue
      wait_async0();      // my async A-tile writes complete
      __syncthreads();    // everyone's tile (A async + B ds-stores) visible
    }
  }

  // Epilogue: C layout = lane holds column ln15, rows v + 8*hf.
#pragma unroll
  for (int sub = 0; sub < 2; ++sub)
#pragma unroll
    for (int t = 0; t < 4; ++t)
#pragma unroll
      for (int v = 0; v < 8; ++v) {
        int row = row0 + wr * 32 + sub * 16 + v + 8 * hf;
        int col = col0 + wc * 64 + t * 16 + ln15;
        if (col < N) {
          float val = acc[sub][t][v];
          if (bias)     val += bias[col];
          if (residual) val += residual[(size_t)row * N + col];
          if (relu)     val = fmaxf(val, 0.f);
          C[(size_t)row * N + col] = val;
        }
      }
}

// ---------------------------------------------------------------------------
// Flash attention (causal). Q,K,V bf16 [B,H,S,DK=64] -> Out fp32 [B,S,D].
// 256 threads = 8 waves; each wave owns 16 query rows (block: 128).
// K-tile staged via async global->LDS; V transposed through VGPRs.
// ---------------------------------------------------------------------------
__global__ __launch_bounds__(256) void flash_attn(
    const u16* __restrict__ Q, const u16* __restrict__ Km,
    const u16* __restrict__ Vm, float* __restrict__ Out) {
  __shared__ u16 Kt[32][72];        // [key][dk]      (natural)
  __shared__ u16 Vt[64][40];        // [dv][key]      (transposed)
  __shared__ u16 Pst[8][16][40];    // per-wave P staging for C->A transpose

  const int tid  = threadIdx.x;
  const int lane = tid & 31, wave = tid >> 5;
  const int hf = lane >> 4, ln15 = lane & 15;
  const int bh = blockIdx.y;
  const int b = bh / NH, hh = bh % NH;
  const int qbase = blockIdx.x * 128 + wave * 16;

  // Q A-fragments for dk chunks {0..31},{32..63}
  Frag aQ[2];
  {
    const u16* qrow = Q + ((size_t)bh * SEQ + qbase + ln15) * DKC;
#pragma unroll
    for (int c = 0; c < 2; ++c) {
      aQ[c].q[0] = *(const u32x4*)(qrow + c * 32 + 8 * hf);
      aQ[c].q[1] = *(const u32x4*)(qrow + c * 32 + 16 + 8 * hf);
    }
  }

  float mrow[8], lrow[8];
  v8f o[4] = {};
#pragma unroll
  for (int v = 0; v < 8; ++v) { mrow[v] = -1e30f; lrow[v] = 0.f; }

  const int ktmax = (blockIdx.x * 128 + 127) >> 5;   // inclusive
  for (int kt = 0; kt <= ktmax; ++kt) {
    __syncthreads();               // previous tile fully consumed
    {
      int r = tid >> 3, cb = (tid & 7) * 8;
      size_t base = ((size_t)bh * SEQ + kt * 32 + r) * DKC + cb;
      async_ld16((unsigned)(size_t)&Kt[r][cb], Km + base);  // K: async -> LDS
      u32x4 vv = *(const u32x4*)(Vm + base);                // V: transpose path
      const u16* vs = (const u16*)&vv;
#pragma unroll
      for (int j = 0; j < 8; ++j) Vt[cb + j][r] = vs[j];
    }
    wait_async0();
    __syncthreads();

    if (kt * 32 <= qbase + 15) {     // wave-uniform: skip fully-masked tiles
      // S = Q @ K^T  (two 16-key column tiles, two dk chunks each)
      Frag bK[2][2];
#pragma unroll
      for (int t = 0; t < 2; ++t) {
        int n = t * 16 + ln15;       // lane's key column
#pragma unroll
        for (int c = 0; c < 2; ++c) {
          bK[t][c].q[0] = *(const u32x4*)&Kt[n][c * 32 + 8 * hf];
          bK[t][c].q[1] = *(const u32x4*)&Kt[n][c * 32 + 16 + 8 * hf];
        }
      }
      v8f sc[2] = {};
#pragma unroll
      for (int t = 0; t < 2; ++t) {
        sc[t] = __builtin_amdgcn_wmma_f32_16x16x32_bf16(
            false, aQ[0].v, false, bK[t][0].v, (short)0, sc[t], false, false);
        sc[t] = __builtin_amdgcn_wmma_f32_16x16x32_bf16(
            false, aQ[1].v, false, bK[t][1].v, (short)0, sc[t], false, false);
      }

      // causal mask + online softmax (row = v + 8*hf; col = key across lanes)
#pragma unroll
      for (int v = 0; v < 8; ++v) {
        int qr   = qbase + 8 * hf + v;
        int key0 = kt * 32 + ln15;
        float s0 = sc[0][v] * 0.125f;           // 1/sqrt(64)
        float s1 = sc[1][v] * 0.125f;
        if (key0 > qr)      s0 = -1e30f;
        if (key0 + 16 > qr) s1 = -1e30f;
        float mx = fmaxf(s0, s1);
        mx = fmaxf(mx, __shfl_xor(mx, 1));
        mx = fmaxf(mx, __shfl_xor(mx, 2));
        mx = fmaxf(mx, __shfl_xor(mx, 4));
        mx = fmaxf(mx, __shfl_xor(mx, 8));
        float mnew = fmaxf(mrow[v], mx);
        float p0 = __expf(s0 - mnew);
        float p1 = __expf(s1 - mnew);
        float ps = p0 + p1;
        ps += __shfl_xor(ps, 1);
        ps += __shfl_xor(ps, 2);
        ps += __shfl_xor(ps, 4);
        ps += __shfl_xor(ps, 8);
        float corr = __expf(mrow[v] - mnew);
        lrow[v] = lrow[v] * corr + ps;
        mrow[v] = mnew;
#pragma unroll
        for (int t = 0; t < 4; ++t) o[t][v] *= corr;
        Pst[wave][v + 8 * hf][ln15]      = f2bf(p0);
        Pst[wave][v + 8 * hf][16 + ln15] = f2bf(p1);
      }

      // P (C-layout) -> A-fragment via per-wave LDS bounce
      Frag aP;
      aP.q[0] = *(const u32x4*)&Pst[wave][ln15][8 * hf];
      aP.q[1] = *(const u32x4*)&Pst[wave][ln15][16 + 8 * hf];

      // O += P @ V   (four 16-wide dv column tiles)
#pragma unroll
      for (int t = 0; t < 4; ++t) {
        Frag bV;
        int n = t * 16 + ln15;
        bV.q[0] = *(const u32x4*)&Vt[n][8 * hf];
        bV.q[1] = *(const u32x4*)&Vt[n][16 + 8 * hf];
        o[t] = __builtin_amdgcn_wmma_f32_16x16x32_bf16(
            false, aP.v, false, bV.v, (short)0, o[t], false, false);
      }
    }
  }

  // normalize + scatter to attn [B,S,D]
#pragma unroll
  for (int t = 0; t < 4; ++t)
#pragma unroll
    for (int v = 0; v < 8; ++v) {
      int row = qbase + 8 * hf + v;
      int col = t * 16 + ln15;
      Out[((size_t)b * SEQ + row) * DM + hh * DKC + col] = o[t][v] / lrow[v];
    }
}

// ---------------------------------------------------------------------------
// Host orchestration
// ---------------------------------------------------------------------------
extern "C" void kernel_launch(void* const* d_in, const int* in_sizes, int n_in,
                              void* d_out, int out_size, void* d_ws, size_t ws_size,
                              hipStream_t stream) {
  const int*   x     = (const int*)  d_in[0];
  const float* emb   = (const float*)d_in[1];
  const float* ln1g  = (const float*)d_in[2];
  const float* ln1b  = (const float*)d_in[3];
  const float* wq    = (const float*)d_in[4];
  const float* wk    = (const float*)d_in[5];
  const float* wv    = (const float*)d_in[6];
  const float* wo    = (const float*)d_in[7];
  const float* ln2g  = (const float*)d_in[8];
  const float* ln2b  = (const float*)d_in[9];
  const float* w1    = (const float*)d_in[10];
  const float* b1    = (const float*)d_in[11];
  const float* w2    = (const float*)d_in[12];
  const float* b2    = (const float*)d_in[13];
  const float* lnfg  = (const float*)d_in[14];
  const float* lnfb  = (const float*)d_in[15];
  const float* headw = (const float*)d_in[16];
  const float* headb = (const float*)d_in[17];

  char* ws = (char*)d_ws;
  size_t off = 0;
  float* h  = (float*)(ws + off); off += (size_t)MROWS * DM * 4;   // residual stream
  u16*   hn = (u16*)  (ws + off); off += (size_t)MROWS * DM * 2;   // LN out / attn bf16
  u16*   qb = (u16*)  (ws + off); off += (size_t)MROWS * DM * 2;   // [B,H,S,DK]
  u16*   kb = (u16*)  (ws + off); off += (size_t)MROWS * DM * 2;
  u16*   vb = (u16*)  (ws + off); off += (size_t)MROWS * DM * 2;
  float* t0 = (float*)(ws + off); off += (size_t)MROWS * FF * 4;   // fp32 scratch
  u16*   t1 = (u16*)  (ws + off); off += (size_t)MROWS * FF * 2;   // bf16 scratch
  u16*   wb = (u16*)  (ws + off);                                  // bf16 weights (<= D*V)

  dim3 blk(256);
  dim3 gD (DM / 128, MROWS / 128);     // N=1024 GEMM grid
  dim3 gF (FF / 128, MROWS / 128);     // N=4096 GEMM grid
  dim3 gV ((NV + 127) / 128, MROWS / 128);

  embed_pe<<<1024, blk, 0, stream>>>(x, emb, h);

  for (int l = 0; l < NL; ++l) {
    // ---- attention sublayer ----
    ln_f32_bf16<<<MROWS, blk, 0, stream>>>(h, ln1g + (size_t)l * DM, ln1b + (size_t)l * DM, hn);

    cvt_f32_bf16<<<2048, blk, 0, stream>>>(wq + (size_t)l * DM * DM, wb, (size_t)DM * DM);
    gemm_bf16<<<gD, blk, 0, stream>>>(hn, wb, t0, nullptr, nullptr, MROWS, DM, DM, 0);
    repack_heads<<<1024, blk, 0, stream>>>(t0, qb);

    cvt_f32_bf16<<<2048, blk, 0, stream>>>(wk + (size_t)l * DM * DM, wb, (size_t)DM * DM);
    gemm_bf16<<<gD, blk, 0, stream>>>(hn, wb, t0, nullptr, nullptr, MROWS, DM, DM, 0);
    repack_heads<<<1024, blk, 0, stream>>>(t0, kb);

    cvt_f32_bf16<<<2048, blk, 0, stream>>>(wv + (size_t)l * DM * DM, wb, (size_t)DM * DM);
    gemm_bf16<<<gD, blk, 0, stream>>>(hn, wb, t0, nullptr, nullptr, MROWS, DM, DM, 0);
    repack_heads<<<1024, blk, 0, stream>>>(t0, vb);

    flash_attn<<<dim3(SEQ / 128, NB * NH), blk, 0, stream>>>(qb, kb, vb, t0);

    cvt_f32_bf16<<<2048, blk, 0, stream>>>(t0, hn, (size_t)MROWS * DM);   // attn -> bf16
    cvt_f32_bf16<<<2048, blk, 0, stream>>>(wo + (size_t)l * DM * DM, wb, (size_t)DM * DM);
    gemm_bf16<<<gD, blk, 0, stream>>>(hn, wb, h, nullptr, h, MROWS, DM, DM, 0);  // h += attn@wo

    // ---- FFN sublayer ----
    ln_f32_bf16<<<MROWS, blk, 0, stream>>>(h, ln2g + (size_t)l * DM, ln2b + (size_t)l * DM, hn);
    cvt_f32_bf16<<<4096, blk, 0, stream>>>(w1 + (size_t)l * DM * FF, wb, (size_t)DM * FF);
    gemm_bf16<<<gF, blk, 0, stream>>>(hn, wb, t0, b1 + (size_t)l * FF, nullptr, MROWS, FF, DM, 1);
    cvt_f32_bf16<<<4096, blk, 0, stream>>>(t0, t1, (size_t)MROWS * FF);
    cvt_f32_bf16<<<4096, blk, 0, stream>>>(w2 + (size_t)l * FF * DM, wb, (size_t)FF * DM);
    gemm_bf16<<<gD, blk, 0, stream>>>(t1, wb, h, b2 + (size_t)l * DM, h, MROWS, DM, FF, 0);
  }

  // ---- final LN + LM head ----
  ln_f32_bf16<<<MROWS, blk, 0, stream>>>(h, lnfg, lnfb, hn);
  cvt_f32_bf16<<<8192, blk, 0, stream>>>(headw, wb, (size_t)DM * NV);
  gemm_bf16<<<gV, blk, 0, stream>>>(hn, wb, (float*)d_out, headb, nullptr, MROWS, NV, DM, 0);
}
